// TransformerBlock_5961414607466
// MI455X (gfx1250) — matrix operand order repaired and checked by reading, not compiled, not run on hardware
//
#include <hip/hip_runtime.h>

typedef _Float16 h16;
typedef __attribute__((ext_vector_type(16))) _Float16 v16h;
typedef __attribute__((ext_vector_type(8)))  float    v8f;

// ---------------------------------------------------------------------------
// WMMA fragment helpers (CDNA5 wave32, V_WMMA_F32_16X16X32_F16)
//  A: row-major [16][lda] halves.  Per ISA 7.12.2 (16-bit A 16x32):
//    lanes 0-15: m=lane,   VGPR j(0..3): K=2j,2j+1 ; VGPR j(4..7): K=16+2(j-4),..
//    lanes 16-31: m=lane-16, same but K offset +8 (j<4) / +8 (j>=4)
//  B: stored N-major [16][ldb] (i.e. column-major of the K x 16 operand):
//    lanes 0-15: n=lane, halves = K 0..7,16..23 ; lanes 16-31: n=lane-16, K 8..15,24..31
// ---------------------------------------------------------------------------
__device__ __forceinline__ v16h load_a_frag(const h16* A, int lda, int lane, int k0) {
    int m  = lane & 15;
    int kb = (lane & 16) ? 8 : 0;
    const h16* row = A + (size_t)m * lda + k0 + kb;
    v16h a;
#pragma unroll
    for (int j = 0; j < 8; ++j) {
        int kk = ((j & 4) ? 16 : 0) + 2 * (j & 3);
        a[2 * j]     = row[kk];
        a[2 * j + 1] = row[kk + 1];
    }
    return a;
}

__device__ __forceinline__ v16h load_b_frag(const h16* Bt, int ldb, int lane, int k0) {
    int n  = lane & 15;
    int kb = (lane & 16) ? 8 : 0;
    const h16* row = Bt + (size_t)n * ldb + k0 + kb;
    v16h b;
#pragma unroll
    for (int i = 0; i < 16; ++i) b[i] = row[(i < 8) ? i : i + 8];
    return b;
}

template <int K>
__device__ __forceinline__ v8f gemm16(const h16* A, int lda, const h16* Bt, int ldb, int lane) {
    v8f acc = {};
    for (int k0 = 0; k0 < K; k0 += 32) {
        v16h a = load_a_frag(A, lda, lane, k0);
        v16h b = load_b_frag(Bt, ldb, lane, k0);
        acc = __builtin_amdgcn_wmma_f32_16x16x32_f16(false, a, false, b, (short)0, acc,
                                                     false, false);
    }
    return acc;
}

// ---------------------------------------------------------------------------
// Prep kernels
// ---------------------------------------------------------------------------
// dst[n*K + k] = src[k*ld + coloff + n]   (fp32 -> fp16, N-major weight)
__global__ void transpose_w(const float* __restrict__ src, h16* __restrict__ dst,
                            int K, int N, int ld, int coloff) {
    int idx = blockIdx.x * blockDim.x + threadIdx.x;
    if (idx >= K * N) return;
    int k = idx / N, n = idx - k * N;
    dst[(size_t)n * K + k] = (h16)src[(size_t)k * ld + coloff + n];
}

// patchify: dst[b][t][kk] with kk = (p1*8+p2)*3+c ; img [8][3][256][256]
__global__ void patchify_k(const float* __restrict__ img, h16* __restrict__ dst) {
    int idx = blockIdx.x * blockDim.x + threadIdx.x;
    if (idx >= 8 * 1024 * 192) return;
    int kk = idx % 192;
    int t  = (idx / 192) & 1023;
    int b  = idx / (192 * 1024);
    int c  = kk % 3;
    int p2 = (kk / 3) & 7;
    int p1 = kk / 24;
    int pr = t >> 5, pc = t & 31;
    int row = pr * 8 + p1, col = pc * 8 + p2;
    dst[idx] = (h16)img[(((size_t)b * 3 + c) * 256 + row) * 256 + col];
}

// ---------------------------------------------------------------------------
// Embed: x = patch @ We + b + sinPE(t)    grid = 2*8*64, block 256
// ---------------------------------------------------------------------------
__global__ void __launch_bounds__(256) embed_kernel(const h16* __restrict__ A0,
                                                    const h16* __restrict__ A1,
                                                    const h16* __restrict__ WeT,
                                                    const float* __restrict__ be,
                                                    h16* __restrict__ x0,
                                                    h16* __restrict__ x1) {
    int bx  = blockIdx.x;
    int img = bx >> 9;
    int rem = bx & 511;
    int b   = rem >> 6;
    int mt  = rem & 63;
    int w = threadIdx.x >> 5, lane = threadIdx.x & 31;
    const h16* A = (img ? A1 : A0) + ((size_t)b * 1024 + mt * 16) * 192;
    h16* xo = (img ? x1 : x0);
    int mhi = (lane & 16) ? 8 : 0;
#pragma unroll
    for (int ii = 0; ii < 2; ++ii) {
        int nt = w * 2 + ii;
        v8f acc = gemm16<192>(A, 192, WeT + (size_t)nt * 16 * 192, 192, lane);
        int n = nt * 16 + (lane & 15);
        float bias = be[n];
        float freq = __powf(10000.0f, -(float)(2 * (n >> 1)) * (1.0f / 256.0f));
        float pepar = (float)(n & 1);  // 0 -> sin, 1 -> cos
#pragma unroll
        for (int r = 0; r < 8; ++r) {
            int t = mt * 16 + r + mhi;
            float ang = (float)t * freq;
            float pe = pepar ? __cosf(ang) : __sinf(ang);
            xo[((size_t)b * 1024 + t) * 256 + n] = (h16)(acc[r] + bias + pe);
        }
    }
}

// ---------------------------------------------------------------------------
// QKV: mode0: k = x0@Wk+bk [b,h,t,32] ; mode1: vT = (x0@Wv+bv)^T [b,h,32,t]
//      mode2: q = (x1@Wq+bq)*scale [b,h,t,32]      grid = 3*8*64
// ---------------------------------------------------------------------------
__global__ void __launch_bounds__(256) qkv_kernel(const h16* __restrict__ x0,
                                                  const h16* __restrict__ x1,
                                                  const h16* __restrict__ WkT,
                                                  const h16* __restrict__ WvT,
                                                  const h16* __restrict__ WqT,
                                                  const float* __restrict__ bkv,
                                                  const float* __restrict__ bqv,
                                                  h16* __restrict__ qd,
                                                  h16* __restrict__ kd,
                                                  h16* __restrict__ vTd) {
    int bx   = blockIdx.x;
    int mode = bx / 512;
    int rem  = bx - mode * 512;
    int b    = rem >> 6;
    int mt   = rem & 63;
    int w = threadIdx.x >> 5, lane = threadIdx.x & 31;
    const h16* src = (mode == 2) ? x1 : x0;
    const h16* Wt  = (mode == 0) ? WkT : (mode == 1) ? WvT : WqT;
    const h16* A = src + ((size_t)b * 1024 + mt * 16) * 256;
    int mhi = (lane & 16) ? 8 : 0;
#pragma unroll
    for (int ii = 0; ii < 2; ++ii) {
        int nt = w * 2 + ii;
        v8f acc = gemm16<256>(A, 256, Wt + (size_t)nt * 16 * 256, 256, lane);
        int n = nt * 16 + (lane & 15);
        int h = n >> 5, dh = n & 31;
        float bias = (mode == 0) ? bkv[n] : (mode == 1) ? bkv[256 + n] : bqv[n];
#pragma unroll
        for (int r = 0; r < 8; ++r) {
            int t = mt * 16 + r + mhi;
            float v = acc[r] + bias;
            size_t bh = (size_t)b * 8 + h;
            if (mode == 0) {
                kd[(bh * 1024 + t) * 32 + dh] = (h16)v;
            } else if (mode == 1) {
                vTd[(bh * 32 + dh) * 1024 + t] = (h16)v;
            } else {
                qd[(bh * 1024 + t) * 32 + dh] = (h16)(v * 0.17677669529663687f);
            }
        }
    }
}

// ---------------------------------------------------------------------------
// Attention: per (b,h, 16-query tile). Scores via WMMA -> LDS, masked,
// in-place exp-softmax (wave reductions), probs@v via WMMA + LDS reduction.
// grid = 8*8*64 = 4096, block 256.
// ---------------------------------------------------------------------------
__global__ void __launch_bounds__(256) attn_kernel(const h16* __restrict__ qd,
                                                   const h16* __restrict__ kd,
                                                   const h16* __restrict__ vTd,
                                                   h16* __restrict__ attn_out) {
    __shared__ h16   S[16][1024];   // scores -> probs (in place), 32 KB
    __shared__ float rsum[16];

    int bx = blockIdx.x;
    int bh = bx >> 6;          // b*8 + h
    int qt = bx & 63;
    int b  = bh >> 3, h = bh & 7;
    int w = threadIdx.x >> 5, lane = threadIdx.x & 31;
    int mhi = (lane & 16) ? 8 : 0;

    // ---- scores: q[16x32] @ k^T, wave w owns key tiles w*8 .. w*8+7
    const h16* qA = qd + ((size_t)bh * 1024 + qt * 16) * 32;
    v16h af = load_a_frag(qA, 32, lane, 0);
    for (int j = 0; j < 8; ++j) {
        int kt = w * 8 + j;
        const h16* kB = kd + ((size_t)bh * 1024 + kt * 16) * 32;  // N-major [key][32]
        v16h bf = load_b_frag(kB, 32, lane, 0);
        v8f acc = {};
        acc = __builtin_amdgcn_wmma_f32_16x16x32_f16(false, af, false, bf, (short)0, acc,
                                                     false, false);
        int ki = kt * 16 + (lane & 15);
        int kr = ki >> 5, kc = ki & 31;
#pragma unroll
        for (int r = 0; r < 8; ++r) {
            int m  = r + mhi;
            int qi = qt * 16 + m;
            int qr = qi >> 5, qc = qi & 31;
            bool neigh = (abs(qr - kr) <= 1) && (abs(qc - kc) <= 1);
            S[m][ki] = (h16)(acc[r] + (neigh ? 0.0f : -100.0f));
        }
    }
    __syncthreads();

    // ---- softmax rows (wave w handles rows 2w, 2w+1); exp in place (fp16 probs)
#pragma unroll
    for (int rr = 0; rr < 2; ++rr) {
        int row = w * 2 + rr;
        float mx = -1e30f;
        for (int i = lane; i < 1024; i += 32) mx = fmaxf(mx, (float)S[row][i]);
#pragma unroll
        for (int off = 16; off > 0; off >>= 1) mx = fmaxf(mx, __shfl_xor(mx, off, 32));
        float sum = 0.0f;
        for (int i = lane; i < 1024; i += 32) {
            float e = __expf((float)S[row][i] - mx);
            S[row][i] = (h16)e;
            sum += e;
        }
#pragma unroll
        for (int off = 16; off > 0; off >>= 1) sum += __shfl_xor(sum, off, 32);
        if (lane == 0) rsum[row] = sum;
    }
    __syncthreads();

    // ---- probs @ v : wave w -> (ntile = w&1 of dh, key quarter = w>>1)
    int ntile = w & 1;
    int part  = w >> 1;
    const h16* vB = vTd + ((size_t)bh * 32 + ntile * 16) * 1024;  // N-major [dh][1024]
    v8f acc = {};
    for (int c = 0; c < 8; ++c) {
        int k0 = part * 256 + c * 32;
        v16h a = load_a_frag(&S[0][0], 1024, lane, k0);
        v16h bf = load_b_frag(vB, 1024, lane, k0);
        acc = __builtin_amdgcn_wmma_f32_16x16x32_f16(false, a, false, bf, (short)0, acc,
                                                     false, false);
    }
    __syncthreads();                 // everyone done reading probs
    float* Sf = (float*)&S[0][0];    // reuse LDS for partials: [wave][m][n] 8*16*16 f32
    {
        int n = lane & 15;
#pragma unroll
        for (int r = 0; r < 8; ++r) Sf[(w * 16 + r + mhi) * 16 + n] = acc[r];
    }
    __syncthreads();

    // ---- cross-wave reduce (4 key quarters) + 1/rowsum, write fp16 [b,t,256]
    for (int idx = threadIdx.x; idx < 512; idx += 256) {
        int m = idx >> 5, dh = idx & 31;
        int nt = dh >> 4, n = dh & 15;
        float s = 0.0f;
#pragma unroll
        for (int p = 0; p < 4; ++p) s += Sf[((p * 2 + nt) * 16 + m) * 16 + n];
        int t = qt * 16 + m;
        attn_out[((size_t)b * 1024 + t) * 256 + h * 32 + dh] = (h16)(s / rsum[m]);
    }
}

// ---------------------------------------------------------------------------
// MLP: fc1 -> exact GELU -> fc2 -> unpatchify scatter.  grid = 8*64, block 256
// ---------------------------------------------------------------------------
__global__ void __launch_bounds__(256) mlp_kernel(const h16* __restrict__ attn,
                                                  const h16* __restrict__ W1T,
                                                  const float* __restrict__ b1,
                                                  const h16* __restrict__ W2T,
                                                  const float* __restrict__ b2,
                                                  float* __restrict__ out) {
    __shared__ h16 Hmid[16][1024];  // 32 KB
    int bx = blockIdx.x;
    int b = bx >> 6, mt = bx & 63;
    int w = threadIdx.x >> 5, lane = threadIdx.x & 31;
    int mhi = (lane & 16) ? 8 : 0;

    const h16* A = attn + ((size_t)b * 1024 + mt * 16) * 256;
    for (int nt = w; nt < 64; nt += 8) {
        v8f acc = gemm16<256>(A, 256, W1T + (size_t)nt * 16 * 256, 256, lane);
        int n = nt * 16 + (lane & 15);
        float bias = b1[n];
#pragma unroll
        for (int r = 0; r < 8; ++r) {
            float x = acc[r] + bias;
            float g = 0.5f * x * (1.0f + erff(x * 0.70710678118654752f));
            Hmid[r + mhi][n] = (h16)g;
        }
    }
    __syncthreads();

    for (int nt = w; nt < 12; nt += 8) {
        v8f acc = gemm16<1024>(&Hmid[0][0], 1024, W2T + (size_t)nt * 16 * 1024, 1024, lane);
        int n = nt * 16 + (lane & 15);
        int c = n % 3, p2 = (n / 3) & 7, p1 = n / 24;
        float bias = b2[n];
#pragma unroll
        for (int r = 0; r < 8; ++r) {
            int t = mt * 16 + r + mhi;
            int pr = t >> 5, pc = t & 31;
            out[(((size_t)b * 3 + c) * 256 + pr * 8 + p1) * 256 + pc * 8 + p2] =
                acc[r] + bias;
        }
    }
}

// ---------------------------------------------------------------------------
// Workspace layout (h16 element offsets)
// ---------------------------------------------------------------------------
static constexpr size_t OFF_WET = 0;                       // 256*192
static constexpr size_t OFF_WKT = OFF_WET + 256 * 192;     // 256*256
static constexpr size_t OFF_WVT = OFF_WKT + 256 * 256;
static constexpr size_t OFF_WQT = OFF_WVT + 256 * 256;
static constexpr size_t OFF_W1T = OFF_WQT + 256 * 256;     // 1024*256
static constexpr size_t OFF_W2T = OFF_W1T + 1024 * 256;    // 192*1024
static constexpr size_t OFF_A0  = OFF_W2T + 192 * 1024;    // 8*1024*192
static constexpr size_t OFF_A1  = OFF_A0 + 8 * 1024 * 192;
static constexpr size_t OFF_X0  = OFF_A1 + 8 * 1024 * 192; // 8*1024*256
static constexpr size_t OFF_X1  = OFF_X0 + 8 * 1024 * 256;
static constexpr size_t OFF_Q   = OFF_X1 + 8 * 1024 * 256; // 8*8*1024*32
static constexpr size_t OFF_K   = OFF_Q + 8 * 8 * 1024 * 32;
static constexpr size_t OFF_VT  = OFF_K + 8 * 8 * 1024 * 32;
static constexpr size_t OFF_ATT = OFF_VT + 8 * 8 * 1024 * 32;

extern "C" void kernel_launch(void* const* d_in, const int* in_sizes, int n_in,
                              void* d_out, int out_size, void* d_ws, size_t ws_size,
                              hipStream_t stream) {
    const float* img0 = (const float*)d_in[0];
    const float* img1 = (const float*)d_in[1];
    const float* W_embed = (const float*)d_in[2];
    const float* b_embed = (const float*)d_in[3];
    const float* W_kv = (const float*)d_in[4];
    const float* b_kv = (const float*)d_in[5];
    const float* W_qv = (const float*)d_in[6];
    const float* b_qv = (const float*)d_in[7];
    const float* W_fc1 = (const float*)d_in[8];
    const float* b_fc1 = (const float*)d_in[9];
    const float* W_fc2 = (const float*)d_in[10];
    const float* b_fc2 = (const float*)d_in[11];
    float* out = (float*)d_out;
    h16* ws = (h16*)d_ws;

    // weight transposes (fp32 -> fp16, N-major)
    transpose_w<<<(192 * 256 + 255) / 256, 256, 0, stream>>>(W_embed, ws + OFF_WET, 192, 256, 256, 0);
    transpose_w<<<(256 * 256 + 255) / 256, 256, 0, stream>>>(W_kv, ws + OFF_WKT, 256, 256, 512, 0);
    transpose_w<<<(256 * 256 + 255) / 256, 256, 0, stream>>>(W_kv, ws + OFF_WVT, 256, 256, 512, 256);
    transpose_w<<<(256 * 256 + 255) / 256, 256, 0, stream>>>(W_qv, ws + OFF_WQT, 256, 256, 512, 0);
    transpose_w<<<(256 * 1024 + 255) / 256, 256, 0, stream>>>(W_fc1, ws + OFF_W1T, 256, 1024, 1024, 0);
    transpose_w<<<(1024 * 192 + 255) / 256, 256, 0, stream>>>(W_fc2, ws + OFF_W2T, 1024, 192, 192, 0);
    // patchify
    patchify_k<<<(8 * 1024 * 192 + 255) / 256, 256, 0, stream>>>(img0, ws + OFF_A0);
    patchify_k<<<(8 * 1024 * 192 + 255) / 256, 256, 0, stream>>>(img1, ws + OFF_A1);
    // embed + PE
    embed_kernel<<<1024, 256, 0, stream>>>(ws + OFF_A0, ws + OFF_A1, ws + OFF_WET, b_embed,
                                           ws + OFF_X0, ws + OFF_X1);
    // q/k/v projections
    qkv_kernel<<<1536, 256, 0, stream>>>(ws + OFF_X0, ws + OFF_X1, ws + OFF_WKT, ws + OFF_WVT,
                                         ws + OFF_WQT, b_kv, b_qv, ws + OFF_Q, ws + OFF_K,
                                         ws + OFF_VT);
    // attention
    attn_kernel<<<4096, 256, 0, stream>>>(ws + OFF_Q, ws + OFF_K, ws + OFF_VT, ws + OFF_ATT);
    // MLP + unpatchify
    mlp_kernel<<<512, 256, 0, stream>>>(ws + OFF_ATT, ws + OFF_W1T, b_fc1, ws + OFF_W2T, b_fc2,
                                        out);
}
